// BahdanauAttnDecoderRNN_3590592659955
// MI455X (gfx1250) — matrix operand order repaired and checked by reading, not compiled, not run on hardware
//
#include <hip/hip_runtime.h>
#include <hip/hip_bf16.h>

#define H   1024
#define VSZ 50257
#define SS  64
#define K2H 2048
#define G3H 3072

typedef __attribute__((ext_vector_type(16))) _Float16 v16h;
typedef __attribute__((ext_vector_type(8)))  float    v8f;

// workspace layout (float offsets)
#define WS_X       0                   // 2048: [embed | context]
#define WS_ENERGY  2048                // 64*1024
#define WS_SCORES  (WS_ENERGY + SS*H)  // 64
#define WS_W       (WS_SCORES + 64)    // 64
#define WS_GI      (WS_W + 64)         // 3072
#define WS_GH      (WS_GI + G3H)       // 3072
#define WS_H0      (WS_GH + G3H)       // 1024
#define WS_H1      (WS_H0 + H)         // 1024
#define WS_Z       (WS_H1 + H)         // 2048: [h1 | context]

// ---------------- embedding gather ----------------
__global__ void embed_kernel(const int* __restrict__ word,
                             const float* __restrict__ emb,
                             float* __restrict__ x) {
    int i = blockIdx.x * 256 + threadIdx.x;
    x[i] = emb[(size_t)word[0] * H + i];
}

// ---------------- attention energy via WMMA (f16 in, f32 acc) ----------------
// energy[s, j] = sum_k cat[s,k] * attn_W[j,k] + attn_b[j]
// cat[s, 0:1024] = hid_top, cat[s, 1024:2048] = enc[s]
// One wave computes one 16x16 tile; tiles: 4 (M=64) x 64 (N=1024).
__global__ void energy_wmma_kernel(const float* __restrict__ attn_W,
                                   const float* __restrict__ attn_b,
                                   const float* __restrict__ hid_top,
                                   const float* __restrict__ enc,
                                   float* __restrict__ energy) {
    int wave = (blockIdx.x * blockDim.x + threadIdx.x) >> 5;   // 0..255
    int lane = threadIdx.x & 31;
    int tm = wave >> 6;          // 0..3
    int tn = wave & 63;          // 0..63
    int m0 = tm * 16, j0 = tn * 16;
    int half = lane >> 4;        // 0: lanes 0-15, 1: lanes 16-31
    int l15  = lane & 15;
    int mA = m0 + l15;           // A row (M) for this lane
    int nB = j0 + l15;           // B column (N) for this lane
    const float* brow = attn_W + (size_t)nB * K2H;   // B[k][n] = attn_W[n*2048 + k]

    v8f c = {};
    for (int phase = 0; phase < 2; ++phase) {
        // phase 0: A comes from hid_top[k]; phase 1: A = enc[mA*H + (k - H)]
        const float* aptr = (phase == 0) ? hid_top : (enc + (size_t)mA * H - H);
        int kstart = phase * H;
        int kend   = kstart + H;
        for (int k0 = kstart; k0 < kend; k0 += 32) {
            v16h a, b;
            // 16-bit A 16x32 layout: lanes<16 hold K = k0+0..7 & k0+16..23;
            // lanes>=16 hold K = k0+8..15 & k0+24..31.
            int kbA = k0 + half * 8;
#pragma unroll
            for (int i = 0; i < 8; ++i) {
                a[i]     = (_Float16)aptr[kbA + i];
                a[8 + i] = (_Float16)aptr[kbA + 16 + i];
            }
            // B 32x16 layout: lanes<16 hold K = k0+0..7 & k0+16..23 (col = lane),
            // lanes>=16 hold K = k0+8..15 & k0+24..31.
            int kbB = k0 + half * 8;
#pragma unroll
            for (int i = 0; i < 8; ++i) {
                b[i]     = (_Float16)brow[kbB + i];
                b[8 + i] = (_Float16)brow[kbB + 16 + i];
            }
            c = __builtin_amdgcn_wmma_f32_16x16x32_f16(
                    /*neg_a=*/false, a, /*neg_b=*/false, b,
                    /*c_mod=*/(short)0, c, /*reuse_a=*/false, /*reuse_b=*/false);
        }
    }
    float bias = attn_b[nB];
#pragma unroll
    for (int r = 0; r < 8; ++r) {
        int M = m0 + r + half * 8;               // f32 C/D layout
        energy[M * H + nB] = c[r] + bias;
    }
}

// ---------------- scores[s] = dot(energy[s,:], attn_v) ----------------
__global__ void scores_kernel(const float* __restrict__ energy,
                              const float* __restrict__ attn_v,
                              float* __restrict__ scores) {
    __shared__ float red[256];
    int s = blockIdx.x;
    float acc = 0.f;
    for (int j = threadIdx.x; j < H; j += 256)
        acc += energy[s * H + j] * attn_v[j];
    red[threadIdx.x] = acc;
    __syncthreads();
    for (int off = 128; off > 0; off >>= 1) {
        if (threadIdx.x < off) red[threadIdx.x] += red[threadIdx.x + off];
        __syncthreads();
    }
    if (threadIdx.x == 0) scores[s] = red[0];
}

// ---------------- softmax over 64 scores ----------------
__global__ void softmax_kernel(const float* __restrict__ scores,
                               float* __restrict__ w,
                               float* __restrict__ attn_out) {
    __shared__ float sv[SS];
    __shared__ float smax, ssum;
    int t = threadIdx.x;
    sv[t] = scores[t];
    __syncthreads();
    if (t == 0) {
        float m = sv[0];
        for (int i = 1; i < SS; ++i) m = fmaxf(m, sv[i]);
        smax = m;
    }
    __syncthreads();
    float e = expf(sv[t] - smax);
    sv[t] = e;
    __syncthreads();
    if (t == 0) {
        float s = 0.f;
        for (int i = 0; i < SS; ++i) s += sv[i];
        ssum = s;
    }
    __syncthreads();
    float wi = e / ssum;
    w[t] = wi;
    attn_out[t] = wi;
}

// ---------------- context[h] = sum_s w[s] * enc[s,h] ----------------
__global__ void context_kernel(const float* __restrict__ enc,
                               const float* __restrict__ w,
                               float* __restrict__ ctx_x,
                               float* __restrict__ ctx_z) {
    int hh = blockIdx.x * 256 + threadIdx.x;
    float acc = 0.f;
#pragma unroll 8
    for (int s = 0; s < SS; ++s) acc += w[s] * enc[s * H + hh];
    ctx_x[hh] = acc;
    ctx_z[hh] = acc;
}

// ---------------- streaming GEMV: out[r] = dot(W[r,:], x) + bias[r] ----------------
// one wave per row; float4 coalesced loads; prefetch next stripe.
__global__ void gemv_kernel(const float* __restrict__ Wm,
                            const float* __restrict__ x,
                            const float* __restrict__ bias,
                            float* __restrict__ out,
                            int rows, int K) {
    int wave = (blockIdx.x * blockDim.x + threadIdx.x) >> 5;
    int lane = threadIdx.x & 31;
    if (wave >= rows) return;
    const float4* wr = (const float4*)(Wm + (size_t)wave * K);
    const float4* xr = (const float4*)x;
    int nvec = K >> 2;
    float acc = 0.f;
    for (int i = lane; i < nvec; i += 32) {
        if (i + 32 < nvec) __builtin_prefetch(wr + i + 32, 0, 1);
        float4 a  = wr[i];
        float4 bx = xr[i];
        acc += a.x * bx.x + a.y * bx.y + a.z * bx.z + a.w * bx.w;
    }
#pragma unroll
    for (int off = 16; off > 0; off >>= 1)
        acc += __shfl_down(acc, off, 32);
    if (lane == 0) out[wave] = acc + (bias ? bias[wave] : 0.f);
}

// ---------------- GRU gates ----------------
__global__ void gru_gate_kernel(const float* __restrict__ gi,
                                const float* __restrict__ gh,
                                const float* __restrict__ hprev,
                                float* __restrict__ h_ws,
                                float* __restrict__ h_out_d,
                                float* __restrict__ h_z /* nullable */) {
    int i = blockIdx.x * 256 + threadIdx.x;
    float r = 1.f / (1.f + expf(-(gi[i] + gh[i])));
    float z = 1.f / (1.f + expf(-(gi[H + i] + gh[H + i])));
    float n = tanhf(gi[2 * H + i] + r * gh[2 * H + i]);
    float h = (1.f - z) * n + z * hprev[i];
    h_ws[i] = h;
    h_out_d[i] = h;
    if (h_z) h_z[i] = h;
}

// ---------------- log-softmax over V=50257, in-place in d_out ----------------
__global__ void logsoftmax_kernel(float* __restrict__ logits) {
    __shared__ float red[1024];
    int t = threadIdx.x;
    float m = -3.402823466e38f;
    for (int i = t; i < VSZ; i += 1024) m = fmaxf(m, logits[i]);
    red[t] = m;
    __syncthreads();
    for (int off = 512; off > 0; off >>= 1) {
        if (t < off) red[t] = fmaxf(red[t], red[t + off]);
        __syncthreads();
    }
    float gmax = red[0];
    __syncthreads();
    float s = 0.f;
    for (int i = t; i < VSZ; i += 1024) s += expf(logits[i] - gmax);
    red[t] = s;
    __syncthreads();
    for (int off = 512; off > 0; off >>= 1) {
        if (t < off) red[t] += red[t + off];
        __syncthreads();
    }
    float lse = gmax + logf(red[0]);
    for (int i = t; i < VSZ; i += 1024) logits[i] -= lse;
}

extern "C" void kernel_launch(void* const* d_in, const int* in_sizes, int n_in,
                              void* d_out, int out_size, void* d_ws, size_t ws_size,
                              hipStream_t stream) {
    const int*   word        = (const int*)  d_in[0];
    const float* last_hidden = (const float*)d_in[1];   // (2,1,1024)
    const float* enc         = (const float*)d_in[2];   // (64,1,1024)
    const float* emb         = (const float*)d_in[3];
    const float* attn_W      = (const float*)d_in[4];
    const float* attn_b      = (const float*)d_in[5];
    const float* attn_v      = (const float*)d_in[6];
    const float* gru_Wih0    = (const float*)d_in[7];
    const float* gru_Whh0    = (const float*)d_in[8];
    const float* gru_bih0    = (const float*)d_in[9];
    const float* gru_bhh0    = (const float*)d_in[10];
    const float* gru_Wih1    = (const float*)d_in[11];
    const float* gru_Whh1    = (const float*)d_in[12];
    const float* gru_bih1    = (const float*)d_in[13];
    const float* gru_bhh1    = (const float*)d_in[14];
    const float* out_W       = (const float*)d_in[15];
    const float* out_b       = (const float*)d_in[16];
    float* ws  = (float*)d_ws;
    float* out = (float*)d_out;   // [logp(V) | h0(H) | h1(H) | attn_w(64)]

    const float* hid0 = last_hidden;       // layer 0 hidden
    const float* hid1 = last_hidden + H;   // layer 1 hidden (== hid_top)

    // 1) embedding -> x[0:1024]
    embed_kernel<<<4, 256, 0, stream>>>(word, emb, ws + WS_X);
    // 2) attention energy (WMMA)
    energy_wmma_kernel<<<64, 128, 0, stream>>>(attn_W, attn_b, hid1, enc, ws + WS_ENERGY);
    // 3) scores + softmax + context (context also into z[1024:2048])
    scores_kernel<<<SS, 256, 0, stream>>>(ws + WS_ENERGY, attn_v, ws + WS_SCORES);
    softmax_kernel<<<1, SS, 0, stream>>>(ws + WS_SCORES, ws + WS_W, out + VSZ + 2 * H);
    context_kernel<<<4, 256, 0, stream>>>(enc, ws + WS_W, ws + WS_X + H, ws + WS_Z + H);
    // 4) GRU layer 0
    gemv_kernel<<<G3H / 8, 256, 0, stream>>>(gru_Wih0, ws + WS_X, gru_bih0, ws + WS_GI, G3H, K2H);
    gemv_kernel<<<G3H / 8, 256, 0, stream>>>(gru_Whh0, hid0, gru_bhh0, ws + WS_GH, G3H, H);
    gru_gate_kernel<<<4, 256, 0, stream>>>(ws + WS_GI, ws + WS_GH, hid0,
                                           ws + WS_H0, out + VSZ, nullptr);
    // 5) GRU layer 1 (h1 also into z[0:1024])
    gemv_kernel<<<G3H / 8, 256, 0, stream>>>(gru_Wih1, ws + WS_H0, gru_bih1, ws + WS_GI, G3H, H);
    gemv_kernel<<<G3H / 8, 256, 0, stream>>>(gru_Whh1, hid1, gru_bhh1, ws + WS_GH, G3H, H);
    gru_gate_kernel<<<4, 256, 0, stream>>>(ws + WS_GI, ws + WS_GH, hid1,
                                           ws + WS_H1, out + VSZ + H, ws + WS_Z);
    // 6) logits (dominant 412 MB stream) -> d_out[0:V], then log-softmax in place
    gemv_kernel<<<(VSZ + 7) / 8, 256, 0, stream>>>(out_W, ws + WS_Z, out_b, out, VSZ, K2H);
    logsoftmax_kernel<<<1, 1024, 0, stream>>>(out);
}
